// SchNetModel_34093450396358
// MI455X (gfx1250) — matrix operand order repaired and checked by reading, not compiled, not run on hardware
//
#include <hip/hip_runtime.h>
#include <hip/hip_bf16.h>

// ---------------------------------------------------------------------------
// SchNet on gfx1250: all 64-wide matmuls via v_wmma_f32_16x16x32_bf16.
// One wave32 owns a 16-row tile (16 edges or 16 nodes); weights pre-converted
// to bf16 (resident in 192MB L2); per-wave LDS staging tiles; scatter via
// global_atomic_add_f32.
// ---------------------------------------------------------------------------

#define N_NODES   100000
#define N_EDGES   1000000
#define N_GRAPHS  256
#define DIM       64
#define NCENT     50
#define GAPF      (5.0f / 49.0f)
#define INV_GAPF  (49.0f / 5.0f)
#define WPB       4   // waves per block (128 threads)

typedef __attribute__((ext_vector_type(16))) __bf16 v16bf;
typedef __attribute__((ext_vector_type(8)))  float  v8f;

union UA { unsigned u[8]; v16bf v; };

// Wave-local LDS fence: lanes of a wave32 are lockstep; DS ops are in-order,
// we just need to stop compiler reordering and drain DScnt.
__device__ __forceinline__ void lds_fence() {
  __builtin_amdgcn_wave_barrier();
  asm volatile("s_wait_dscnt 0" ::: "memory");
  __builtin_amdgcn_wave_barrier();
}

// softplus with beta=0.5: 2*log1p(exp(0.5x)), numerically stable
__device__ __forceinline__ float sp05(float x) {
  float t = 0.5f * x;
  return 2.0f * (fmaxf(t, 0.0f) + log1pf(__expf(-fabsf(t))));
}
// softplus beta=1 (readout ShiftSoftplus applies -ln2 outside)
__device__ __forceinline__ float sp1(float x) {
  return fmaxf(x, 0.0f) + log1pf(__expf(-fabsf(x)));
}

// A operand: 16x32 bf16 tile slice from LDS tile stored row-major [16][64]
// ISA layout: lanes 0-15 = rows, VGPR j<4 -> K pair {half*8+2j}, j>=4 -> {16+half*8+2(j-4)}
__device__ __forceinline__ v16bf load_A(const unsigned* lds_u, int kstep, int lane) {
  int row  = lane & 15;
  int half = (lane >> 4) & 1;
  UA a;
#pragma unroll
  for (int j = 0; j < 8; ++j) {
    int k = kstep * 32 + half * 8 + ((j < 4) ? (2 * j) : (16 + 2 * (j - 4)));
    a.u[j] = lds_u[row * 32 + (k >> 1)];
  }
  return a.v;
}

// B operand: 32x16 bf16 slice of W^T taken directly from row-major W[64][64] bf16.
// B(k,n) = W[n][k]; same K layout as A: VGPR j<4 -> K pair {half*8+2j}, j>=4 -> {16+half*8+2(j-4)}.
__device__ __forceinline__ v16bf load_B(const unsigned* w_u, int ntile, int kstep, int lane) {
  int n    = ntile * 16 + (lane & 15);
  int koff = kstep * 32 + ((lane >> 4) & 1) * 8;
  UA b;
#pragma unroll
  for (int j = 0; j < 8; ++j) b.u[j] = w_u[n * 32 + (koff >> 1) + ((j < 4) ? j : (j + 4))];
  return b.v;
}

// One 16x16 output tile of X[16,64] @ W^T[64,64] (+bias), K=64 -> 2 WMMA steps
__device__ __forceinline__ v8f mm_tile(const unsigned* tA_u, const unsigned* w_u,
                                       const float* bias, int ntile, int lane) {
  float bb = bias ? bias[ntile * 16 + (lane & 15)] : 0.0f;
  v8f acc = {bb, bb, bb, bb, bb, bb, bb, bb};
#pragma unroll
  for (int ks = 0; ks < 2; ++ks) {
    v16bf a = load_A(tA_u, ks, lane);
    v16bf b = load_B(w_u, ntile, ks, lane);
    acc = __builtin_amdgcn_wmma_f32_16x16x32_bf16(false, a, false, b,
                                                  (short)0, acc, false, false);
  }
  return acc;
}

// ---------------------------------------------------------------------------
// Weight prep: f32 -> bf16 once. Layout: 16 matrices of 64x64 bf16.
// m = i*5 + {0:W1, 1:cfW1 (K padded 50->64), 2:cfW2, 3:W2, 4:W3}; m==15: d1W
// ---------------------------------------------------------------------------
__global__ void prep_weights_kernel(const float* __restrict__ W1,
                                    const float* __restrict__ cfW1,
                                    const float* __restrict__ cfW2,
                                    const float* __restrict__ W2,
                                    const float* __restrict__ W3,
                                    const float* __restrict__ d1W,
                                    __bf16* __restrict__ out) {
  int idx = blockIdx.x * 256 + threadIdx.x;
  if (idx >= 16 * 4096) return;
  int m = idx >> 12, r = (idx >> 6) & 63, c = idx & 63;
  float v;
  if (m == 15) v = d1W[r * 64 + c];
  else {
    int i = m / 5, w = m % 5;
    if      (w == 0) v = W1  [(i * 64 + r) * 64 + c];
    else if (w == 1) v = (c < NCENT) ? cfW1[(i * 64 + r) * NCENT + c] : 0.0f;
    else if (w == 2) v = cfW2[(i * 64 + r) * 64 + c];
    else if (w == 3) v = W2  [(i * 64 + r) * 64 + c];
    else             v = W3  [(i * 64 + r) * 64 + c];
  }
  out[idx] = (__bf16)v;
}

// node = emb[node_type], float4-vectorized
__global__ void embed_kernel(const int* __restrict__ node_type,
                             const float* __restrict__ emb,
                             float* __restrict__ node) {
  int idx = blockIdx.x * 256 + threadIdx.x;
  if (idx >= N_NODES * 16) return;
  int n = idx >> 4, q = idx & 15;
  int t = node_type[n];
  ((float4*)node)[n * 16 + q] = ((const float4*)emb)[t * 16 + q];
}

// new_node = node @ W1^T (no bias)
__global__ __launch_bounds__(128) void node_lin1_kernel(
    const float* __restrict__ node, const unsigned* __restrict__ bW1,
    float* __restrict__ new_node, int n_tiles) {
  __shared__ __attribute__((aligned(16))) __bf16 sA[WPB][1024];
  int wave = threadIdx.x >> 5, lane = threadIdx.x & 31;
  int tile = blockIdx.x * WPB + wave;
  if (tile >= n_tiles) return;
  __bf16* tA = sA[wave];
  int n0 = tile * 16;
#pragma unroll
  for (int j = 0; j < 32; ++j) {
    int flat = lane * 32 + j;
    tA[flat] = (__bf16)node[n0 * 64 + flat];
  }
  lds_fence();
  const unsigned* tA_u = (const unsigned*)tA;
  int half = lane >> 4, nn = lane & 15;
#pragma unroll
  for (int t = 0; t < 4; ++t) {
    v8f acc = mm_tile(tA_u, bW1, nullptr, t, lane);
#pragma unroll
    for (int r = 0; r < 8; ++r)
      new_node[(n0 + r + half * 8) * 64 + t * 16 + nn] = acc[r];
  }
}

// Edge kernel: rbf -> WMMA(cfW1)+softplus -> WMMA(cfW2)+bias -> * new_node[src]
// -> atomicAdd into cf[dst]
struct __attribute__((aligned(16))) EdgeLds {
  __bf16 tA[1024];   // rbf (bf16, 16x64, K padded)
  __bf16 tB[1024];   // h   (bf16, 16x64)
  float  tG[1024];   // gathered new_node[src] (f32, 16x64)
  int    tS[16];
  int    tD[16];
};

__global__ __launch_bounds__(128) void edge_conv_kernel(
    const float* __restrict__ distance, const int* __restrict__ src,
    const int* __restrict__ dst, const float* __restrict__ new_node,
    float* __restrict__ cf,
    const unsigned* __restrict__ bCf1, const unsigned* __restrict__ bCf2,
    const float* __restrict__ cfb1, const float* __restrict__ cfb2,
    int n_tiles) {
  __shared__ EdgeLds lds[WPB];
  int wave = threadIdx.x >> 5, lane = threadIdx.x & 31;
  int tile = blockIdx.x * WPB + wave;
  if (tile >= n_tiles) return;
  EdgeLds& L = lds[wave];
  int e0 = tile * 16;

  if (lane < 16) { L.tS[lane] = src[e0 + lane]; L.tD[lane] = dst[e0 + lane]; }
  { // rbf: 2 lanes per edge, 32 centers each; pad K 50->64 with zeros
    int e = lane >> 1;
    float d = distance[e0 + e];
    int c0 = (lane & 1) * 32;
#pragma unroll
    for (int j = 0; j < 32; ++j) {
      int c = c0 + j;
      float diff = d - (float)c * GAPF;
      float v = (c < NCENT) ? __expf(-diff * diff * INV_GAPF) : 0.0f;
      L.tA[e * 64 + c] = (__bf16)v;
    }
  }
  lds_fence();

  // gather new_node[src] rows (float4) while matmul1 operands come from LDS
#pragma unroll
  for (int q = 0; q < 8; ++q) {
    int flat4 = lane * 8 + q;           // 0..255 float4s
    int e = flat4 >> 4, c4 = flat4 & 15;
    int s = L.tS[e];
    ((float4*)L.tG)[flat4] = ((const float4*)new_node)[s * 16 + c4];
  }

  const unsigned* tA_u = (const unsigned*)L.tA;
  int half = lane >> 4, nn = lane & 15;
#pragma unroll
  for (int t = 0; t < 4; ++t) {          // h = softplus05(rbf @ cfW1^T + b)
    v8f acc = mm_tile(tA_u, bCf1, cfb1, t, lane);
#pragma unroll
    for (int r = 0; r < 8; ++r)
      L.tB[(r + half * 8) * 64 + t * 16 + nn] = (__bf16)sp05(acc[r]);
  }
  lds_fence();

  const unsigned* tB_u = (const unsigned*)L.tB;
#pragma unroll
  for (int t = 0; t < 4; ++t) {          // h2 = h @ cfW2^T + b; msg; scatter
    v8f acc = mm_tile(tB_u, bCf2, cfb2, t, lane);
#pragma unroll
    for (int r = 0; r < 8; ++r) {
      int row = r + half * 8;
      int col = t * 16 + nn;
      float msg = acc[r] * L.tG[row * 64 + col];
      atomicAdd(&cf[L.tD[row] * 64 + col], msg);
    }
  }
}

// a = softplus05(cf @ W2^T + b2); node += a @ W3^T + b3
__global__ __launch_bounds__(128) void node_update_kernel(
    const float* __restrict__ cf, const unsigned* __restrict__ bW2,
    const unsigned* __restrict__ bW3, const float* __restrict__ b2,
    const float* __restrict__ b3, float* __restrict__ node, int n_tiles) {
  __shared__ __attribute__((aligned(16))) __bf16 sA[WPB][1024];
  __shared__ __attribute__((aligned(16))) __bf16 sB[WPB][1024];
  int wave = threadIdx.x >> 5, lane = threadIdx.x & 31;
  int tile = blockIdx.x * WPB + wave;
  if (tile >= n_tiles) return;
  __bf16* tA = sA[wave];
  __bf16* tB = sB[wave];
  int n0 = tile * 16;
#pragma unroll
  for (int j = 0; j < 32; ++j) {
    int flat = lane * 32 + j;
    tA[flat] = (__bf16)cf[n0 * 64 + flat];
  }
  lds_fence();
  const unsigned* tA_u = (const unsigned*)tA;
  int half = lane >> 4, nn = lane & 15;
#pragma unroll
  for (int t = 0; t < 4; ++t) {
    v8f acc = mm_tile(tA_u, bW2, b2, t, lane);
#pragma unroll
    for (int r = 0; r < 8; ++r)
      tB[(r + half * 8) * 64 + t * 16 + nn] = (__bf16)sp05(acc[r]);
  }
  lds_fence();
  const unsigned* tB_u = (const unsigned*)tB;
#pragma unroll
  for (int t = 0; t < 4; ++t) {
    v8f acc = mm_tile(tB_u, bW3, b3, t, lane);
#pragma unroll
    for (int r = 0; r < 8; ++r)
      node[(n0 + r + half * 8) * 64 + t * 16 + nn] += acc[r];
  }
}

// atom = softplus1(node @ d1W^T + d1b) - ln2 ; res = atom @ d2W^T + d2b ;
// per-graph atomic sum + count
__global__ __launch_bounds__(128) void readout_kernel(
    const float* __restrict__ node, const unsigned* __restrict__ bD1,
    const float* __restrict__ d1b, const float* __restrict__ d2W,
    const float* __restrict__ d2b, const int* __restrict__ graph_ids,
    float* __restrict__ gsum, float* __restrict__ gcnt, int n_tiles) {
  __shared__ __attribute__((aligned(16))) __bf16 sA[WPB][1024];
  __shared__ __attribute__((aligned(16))) float  sF[WPB][1024];
  int wave = threadIdx.x >> 5, lane = threadIdx.x & 31;
  int tile = blockIdx.x * WPB + wave;
  if (tile >= n_tiles) return;
  __bf16* tA = sA[wave];
  float*  tF = sF[wave];
  int n0 = tile * 16;
#pragma unroll
  for (int j = 0; j < 32; ++j) {
    int flat = lane * 32 + j;
    tA[flat] = (__bf16)node[n0 * 64 + flat];
  }
  lds_fence();
  const unsigned* tA_u = (const unsigned*)tA;
  int half = lane >> 4, nn = lane & 15;
#pragma unroll
  for (int t = 0; t < 4; ++t) {
    v8f acc = mm_tile(tA_u, bD1, d1b, t, lane);
#pragma unroll
    for (int r = 0; r < 8; ++r)
      tF[(r + half * 8) * 64 + t * 16 + nn] = sp1(acc[r]) - 0.6931471805599453f;
  }
  lds_fence();
  if (lane < 16) {
    float s = 0.0f;
#pragma unroll 8
    for (int c = 0; c < 64; ++c) s += tF[lane * 64 + c] * d2W[c];
    s += d2b[0];
    int g = graph_ids[n0 + lane];
    atomicAdd(&gsum[g], s);
    atomicAdd(&gcnt[g], 1.0f);
  }
}

__global__ void finalize_kernel(const float* __restrict__ gsum,
                                const float* __restrict__ gcnt,
                                float* __restrict__ out) {
  int g = threadIdx.x;
  if (g < N_GRAPHS) out[g] = gsum[g] / fmaxf(gcnt[g], 1.0f);
}

// ---------------------------------------------------------------------------
extern "C" void kernel_launch(void* const* d_in, const int* in_sizes, int n_in,
                              void* d_out, int out_size, void* d_ws, size_t ws_size,
                              hipStream_t stream) {
  const int*   node_type = (const int*)d_in[0];
  const int*   src       = (const int*)d_in[1];
  const int*   dst       = (const int*)d_in[2];
  const float* distance  = (const float*)d_in[3];
  const int*   graph_ids = (const int*)d_in[4];
  const float* emb       = (const float*)d_in[5];
  const float* W1        = (const float*)d_in[6];
  const float* cfW1      = (const float*)d_in[7];
  const float* cfb1      = (const float*)d_in[8];
  const float* cfW2      = (const float*)d_in[9];
  const float* cfb2      = (const float*)d_in[10];
  const float* W2        = (const float*)d_in[11];
  const float* b2        = (const float*)d_in[12];
  const float* W3        = (const float*)d_in[13];
  const float* b3        = (const float*)d_in[14];
  const float* d1W       = (const float*)d_in[15];
  const float* d1b       = (const float*)d_in[16];
  const float* d2W       = (const float*)d_in[17];
  const float* d2b       = (const float*)d_in[18];
  float* out = (float*)d_out;

  // workspace layout (all 16B aligned)
  const size_t featBytes = (size_t)N_NODES * 64 * sizeof(float);   // 25.6 MB
  char* ws = (char*)d_ws;
  float*  node     = (float*)(ws);
  float*  new_node = (float*)(ws + featBytes);
  float*  cf       = (float*)(ws + 2 * featBytes);
  __bf16* bfw      = (__bf16*)(ws + 3 * featBytes);                // 128 KB
  float*  gsum     = (float*)(ws + 3 * featBytes + 131072);
  float*  gcnt     = gsum + N_GRAPHS;

  const int node_tiles = N_NODES / 16;   // 6250 (exact)
  const int edge_tiles = N_EDGES / 16;   // 62500 (exact)
  const int nb_node = (node_tiles + WPB - 1) / WPB;
  const int nb_edge = (edge_tiles + WPB - 1) / WPB;

  prep_weights_kernel<<<256, 256, 0, stream>>>(W1, cfW1, cfW2, W2, W3, d1W, bfw);
  embed_kernel<<<(N_NODES * 16 + 255) / 256, 256, 0, stream>>>(node_type, emb, node);

  const unsigned* bfw_u = (const unsigned*)bfw;   // 2048 uints per 64x64 bf16 matrix
  for (int i = 0; i < 3; ++i) {
    const unsigned* bW1  = bfw_u + (size_t)(i * 5 + 0) * 2048;
    const unsigned* bCf1 = bfw_u + (size_t)(i * 5 + 1) * 2048;
    const unsigned* bCf2 = bfw_u + (size_t)(i * 5 + 2) * 2048;
    const unsigned* bW2  = bfw_u + (size_t)(i * 5 + 3) * 2048;
    const unsigned* bW3  = bfw_u + (size_t)(i * 5 + 4) * 2048;

    hipMemsetAsync(cf, 0, featBytes, stream);
    node_lin1_kernel<<<nb_node, 128, 0, stream>>>(node, bW1, new_node, node_tiles);
    edge_conv_kernel<<<nb_edge, 128, 0, stream>>>(distance, src, dst, new_node, cf,
                                                  bCf1, bCf2, cfb1 + i * 64,
                                                  cfb2 + i * 64, edge_tiles);
    node_update_kernel<<<nb_node, 128, 0, stream>>>(cf, bW2, bW3, b2 + i * 64,
                                                    b3 + i * 64, node, node_tiles);
  }

  hipMemsetAsync(gsum, 0, 2 * N_GRAPHS * sizeof(float), stream);
  const unsigned* bD1 = bfw_u + (size_t)15 * 2048;
  readout_kernel<<<nb_node, 128, 0, stream>>>(node, bD1, d1b, d2W, d2b, graph_ids,
                                              gsum, gcnt, node_tiles);
  finalize_kernel<<<1, 256, 0, stream>>>(gsum, gcnt, out);
}
